// RBFLayer_75771813037022
// MI455X (gfx1250) — matrix operand order repaired and checked by reading, not compiled, not run on hardware
//
#include <hip/hip_runtime.h>

// RBF layer: out = exp(-GAMMA * (||x||^2 - 2 x@mu + ||mu||^2))
// x: (32768, 512) f32 row-major, mu: (512, 1024) f32 row-major, out: (32768, 1024) f32.
// bf16 WMMA GEMM (fp32 accumulate) + exact fp32 norms fused in one kernel.
// Double-buffered LDS, depth-2 global->LDS software pipeline, NT output stores.

#define GAMMA 0.1f

typedef __attribute__((ext_vector_type(16))) __bf16 bf16x16;
typedef __attribute__((ext_vector_type(4)))  __bf16 bf16x4;
typedef __attribute__((ext_vector_type(2)))  __bf16 bf16x2;
typedef __attribute__((ext_vector_type(8)))  float  f32x8;
typedef __attribute__((ext_vector_type(4)))  float  f32x4;
typedef __attribute__((ext_vector_type(2)))  float  f32x2;

union Frag {
  bf16x16 v;
  uint4   u[2];
};

__global__ __launch_bounds__(256) void rbf_wmma_kernel(
    const float* __restrict__ x, const float* __restrict__ mu,
    float* __restrict__ out) {
  constexpr int D  = 512;
  constexpr int U  = 1024;
  constexpr int KC = 32;        // K per WMMA chunk
  constexpr int NKC = D / KC;   // 16 chunks

  // Double-buffered bf16 tiles; fragment reads are 2x ds_load_b128 per lane.
  __shared__ __align__(16) unsigned short ldsA[2][128 * KC]; // [buf][row][k]
  __shared__ __align__(16) unsigned short ldsB[2][128 * KC]; // [buf][col][k]
  __shared__ float sqx[128];
  __shared__ float sqmu[128];
  __shared__ float sqpart[256];

  const int t    = threadIdx.x;
  const int lane = t & 31;
  const int wave = t >> 5;
  const int waveM = wave >> 2;   // 0..1, 64 rows each
  const int waveN = wave & 3;    // 0..3, 32 cols each

  const size_t blockM = (size_t)blockIdx.y * 128;
  const size_t blockN = (size_t)blockIdx.x * 128;

  // A tile loader: 128 rows x 32 k fp32; each thread loads 16 floats of one row.
  const int arow = t >> 1;
  const int acol = (t & 1) * 16;
  const float* aptr = x + (blockM + (size_t)arow) * D + acol;

  // B tile loader: 32 k x 128 n fp32; thread owns column bn, 16 consecutive k.
  const int bn  = t & 127;
  const int bkh = t >> 7;        // k half (0/1)
  const float* bptr = mu + (size_t)(bkh * 16) * U + blockN + bn;

  f32x4 aReg[4];
  float bReg[16];
  float sqa = 0.f, sqb = 0.f;

  auto loadA = [&](int kc) {
    const f32x4* p = (const f32x4*)(aptr + kc * KC);
#pragma unroll
    for (int j = 0; j < 4; ++j) aReg[j] = p[j];
  };
  auto loadB = [&](int kc) {
    const float* p = bptr + (size_t)kc * KC * U;
#pragma unroll
    for (int i = 0; i < 16; ++i) bReg[i] = p[(size_t)i * U];
  };

  // Convert register-staged chunk to bf16 (native fptrunc -> v_cvt_*bf16),
  // write to LDS buffer, accumulate exact fp32 squared norms.
  auto stage = [&](int buf) {
    union { bf16x4 h[4]; uint4 u[2]; } pa;
#pragma unroll
    for (int j = 0; j < 4; ++j) {
      f32x4 v = aReg[j];
      sqa += v.x * v.x + v.y * v.y + v.z * v.z + v.w * v.w;
      pa.h[j] = __builtin_convertvector(v, bf16x4);
    }
    *(uint4*)&ldsA[buf][arow * KC + acol]     = pa.u[0];
    *(uint4*)&ldsA[buf][arow * KC + acol + 8] = pa.u[1];

    union { bf16x2 h[8]; uint4 u[2]; } pb;
#pragma unroll
    for (int i = 0; i < 8; ++i) {
      f32x2 v = { bReg[2 * i], bReg[2 * i + 1] };
      sqb += v.x * v.x + v.y * v.y;
      pb.h[i] = __builtin_convertvector(v, bf16x2);
    }
    *(uint4*)&ldsB[buf][bn * KC + bkh * 16]     = pb.u[0];
    *(uint4*)&ldsB[buf][bn * KC + bkh * 16 + 8] = pb.u[1];
  };

  f32x8 acc[4][2] = {};

  // Pipeline prologue: chunk 0 staged to buf0; chunk 1 in registers.
  loadA(0); loadB(0);
  stage(0);
  loadA(1); loadB(1);

  for (int kc = 0; kc < NKC; ++kc) {
    __syncthreads();               // buf[kc&1] stores visible to all waves
    const int cur = kc & 1;

    // ---- fragment loads (ISA 7.12.2 layouts) ----
    Frag aF[4], bF[2];
    const int ka = (lane >> 4) * 8;    // A: lanes 16-31 start at K=8 (elements)
    const int kb = (lane >> 4) * 8;    // B: lanes 16-31 start at K=8 (elements)
#pragma unroll
    for (int mi = 0; mi < 4; ++mi) {
      int r = waveM * 64 + mi * 16 + (lane & 15);
      aF[mi].u[0] = *(const uint4*)&ldsA[cur][r * KC + ka];       // K 0-7 / 8-15
      aF[mi].u[1] = *(const uint4*)&ldsA[cur][r * KC + ka + 16];  // K 16-23 / 24-31
    }
#pragma unroll
    for (int ni = 0; ni < 2; ++ni) {
      int c = waveN * 32 + ni * 16 + (lane & 15);
      bF[ni].u[0] = *(const uint4*)&ldsB[cur][c * KC + kb];       // K 0-7 / 8-15
      bF[ni].u[1] = *(const uint4*)&ldsB[cur][c * KC + kb + 16];  // K 16-23 / 24-31
    }

#pragma unroll
    for (int mi = 0; mi < 4; ++mi)
#pragma unroll
      for (int ni = 0; ni < 2; ++ni)
        acc[mi][ni] = __builtin_amdgcn_wmma_f32_16x16x32_bf16(
            false, aF[mi].v, false, bF[ni].v, (short)0, acc[mi][ni],
            false, false);

    // Stage chunk kc+1 (already in registers) into the other buffer; these
    // cvt/ds ops co-execute with the WMMAs above. Then start global loads
    // for chunk kc+2 (a full chunk of latency hiding).
    if (kc + 1 < NKC) stage(cur ^ 1);
    if (kc + 2 < NKC) { loadA(kc + 2); loadB(kc + 2); }
  }

  // ---- finalize norms ----
  float sqa_tot = sqa + __shfl_xor(sqa, 1, 32);  // pair of threads per row
  if ((t & 1) == 0) sqx[arow] = sqa_tot;
  sqpart[t] = sqb;                                // two k-halves per column
  __syncthreads();
  if (t < 128) sqmu[t] = sqpart[t] + sqpart[t + 128];
  __syncthreads();

  // ---- epilogue: e = ||x||^2 - 2*cross + ||mu||^2 ; out = exp(-g*e) ----
#pragma unroll
  for (int mi = 0; mi < 4; ++mi) {
#pragma unroll
    for (int ni = 0; ni < 2; ++ni) {
      const int lr0 = waveM * 64 + mi * 16 + (lane >> 4) * 8;
      const int lc  = waveN * 32 + ni * 16 + (lane & 15);
      const float sm = sqmu[lc];
      float* op = out + (blockM + lr0) * U + blockN + lc;
#pragma unroll
      for (int v = 0; v < 8; ++v) {
        float e = sqx[lr0 + v] - 2.0f * acc[mi][ni][v] + sm;
        __builtin_nontemporal_store(__expf(-GAMMA * e), &op[(size_t)v * U]);
      }
    }
  }
}

extern "C" void kernel_launch(void* const* d_in, const int* in_sizes, int n_in,
                              void* d_out, int out_size, void* d_ws, size_t ws_size,
                              hipStream_t stream) {
  (void)n_in; (void)out_size; (void)d_ws; (void)ws_size;
  const float* x  = (const float*)d_in[0];
  const float* mu = (const float*)d_in[1];
  float* out = (float*)d_out;

  const int D = 512, U = 1024;
  const int N = in_sizes[0] / D;           // 32768
  dim3 grid(U / 128, N / 128);             // (8, 256)
  rbf_wmma_kernel<<<grid, 256, 0, stream>>>(x, mu, out);
}
